// LSTM_34213709480663
// MI455X (gfx1250) — matrix operand order repaired and checked by reading, not compiled, not run on hardware
//
#include <hip/hip_runtime.h>
#include <stdint.h>

// Problem constants (from reference)
#define B_   256
#define T_   1024
#define IN_  64
#define Z_   512
#define S_   2
#define H_   510
#define KDIM 576            // S_ + IN_ + H_  (= 18 * 32, exact)
#define KT   18             // K tiles of 32
#define WAVES 32            // waves per block (1024 threads)
#define NTILE 4             // N-tiles per wave: one per gate (i,f,g,o)
#define MT   16             // batch rows per block (WMMA M)
#define NBLK (B_ / MT)      // 16 independent recurrences
#define MHU_ 1.5f
#define WTILE_BYTES (NTILE * 32 * 16 * 2)   // 4096B of weights per wave per K-tile

typedef __attribute__((ext_vector_type(16))) __bf16 v16bf;
typedef __attribute__((ext_vector_type(8)))  float  v8f;
typedef unsigned int u32x4 __attribute__((ext_vector_type(4)));
typedef float f32x4 __attribute__((ext_vector_type(4)));

union Frag { v16bf v; u32x4 q[2]; };

__device__ __forceinline__ unsigned short f2bf(float f) {
  union { float f; unsigned int u; } x; x.f = f;
  unsigned int u = x.u;
  unsigned int r = u + 0x7FFFu + ((u >> 16) & 1u);   // round-to-nearest-even
  return (unsigned short)(r >> 16);
}

__device__ __forceinline__ float sigmoidf(float x) {
  return 1.0f / (1.0f + __expf(-x));
}

// ---------------------------------------------------------------------------
// Pack [W_w | U_w] (f32, row r = gate*510 + j, col K over [x(2) inp(64) h(510)])
// into bf16 fragment-native layout:
//   Wc[(((w*KT + kb)*NTILE + t)*32 + lane)*16 + e]
// lane<16 holds K = kb*32 + e for N=lane;  lane>=16 holds K = kb*32+16+e.
// Wave w / tile t covers gate t, hidden units j = w*16 + (lane&15).
// ---------------------------------------------------------------------------
__global__ __launch_bounds__(256) void ctrnn_pack_weights(
    const float* __restrict__ Ww, const float* __restrict__ Uw,
    unsigned short* __restrict__ Wc) {
  int gid = blockIdx.x * blockDim.x + threadIdx.x;
  const int NTOT = WAVES * KT * NTILE * 32 * 16;
  if (gid >= NTOT) return;
  int e    = gid & 15;
  int lane = (gid >> 4) & 31;
  int t    = (gid >> 9) & 3;
  int wk   = gid >> 11;
  int kb   = wk % KT;
  int w    = wk / KT;
  int q = lane & 15;
  int j = w * 16 + q;
  int K = kb * 32 + ((lane & 16) ? 16 : 0) + e;
  float val = 0.0f;
  if (j < H_) {
    int r = t * H_ + j;
    if (K < S_ + IN_) val = Ww[(size_t)r * (S_ + IN_) + K];
    else              val = Uw[(size_t)r * H_ + (K - (S_ + IN_))];
  }
  Wc[gid] = f2bf(val);
}

// ---------------------------------------------------------------------------
// Persistent recurrence kernel: one block = 16 batch rows, all T steps.
// hbuf row m holds the full output row [x0, x1, h0..h509] (f32) for row m.
// u_buf h-region is maintained in bf16 directly by the GEMM lanes.
// ---------------------------------------------------------------------------
__global__ __launch_bounds__(1024) void ctrnn_persistent(
    const float* __restrict__ inp,   const float* __restrict__ tau,
    const float* __restrict__ z0,    const float* __restrict__ cz0,
    const float* __restrict__ Wb,
    const float* __restrict__ alpha_w, const float* __restrict__ alpha_b,
    const float* __restrict__ hx_w,    const float* __restrict__ hx_b,
    const unsigned short* __restrict__ Wc,
    float* __restrict__ out) {

  __shared__ unsigned short u_buf[MT * KDIM];   // 18 KB  bf16 [x | inp | h]
  __shared__ float hbuf[MT][Z_];                // 32 KB  [x(2) | h(510)]

  const int tid  = threadIdx.x;
  const int wave = tid >> 5;
  const int lane = tid & 31;
  const int b0   = blockIdx.x * MT;
  const int q    = lane & 15;          // WMMA N / A-row index
  const int hi   = lane >> 4;          // half-wave select
  const int j    = wave * 16 + q;      // hidden unit owned by this lane
  const bool jv  = (j < H_);

  float* zfin  = out + (size_t)B_ * T_ * Z_;
  float* czfin = zfin + (size_t)B_ * Z_;

  // ---- load initial state ----
  for (int idx = tid; idx < MT * Z_; idx += 1024) {     // full z0 row -> hbuf
    int m = idx >> 9, k = idx & (Z_ - 1);
    hbuf[m][k] = z0[(size_t)(b0 + m) * Z_ + k];
  }
  for (int idx = tid; idx < MT * H_; idx += 1024) {     // seed u_buf h-region
    int m = idx / H_, k = idx % H_;
    u_buf[m * KDIM + S_ + IN_ + k] = f2bf(z0[(size_t)(b0 + m) * Z_ + S_ + k]);
  }
  if (tid < MT * S_) {
    int m = tid >> 1, s = tid & 1;
    czfin[(size_t)(b0 + m) * Z_ + s] = cz0[(size_t)(b0 + m) * Z_ + s]; // cx passthrough
  }
  float c[8];                                   // register-resident cell state
  #pragma unroll
  for (int v = 0; v < 8; ++v) {
    int m = hi * 8 + v;
    c[v] = jv ? cz0[(size_t)(b0 + m) * Z_ + S_ + j] : 0.0f;
  }
  float bias[4];                                // loop-invariant gate biases
  #pragma unroll
  for (int t = 0; t < 4; ++t) bias[t] = jv ? Wb[t * H_ + j] : 0.0f;

  __syncthreads();

  for (int step = 0; step < T_; ++step) {
    // (a) stage x + inp_t regions of u (h-region is maintained in place):
    // exactly one input element per thread, no divmods.
    {
      int m = tid >> 6, k = tid & 63;            // MT*IN_ == 1024
      u_buf[m * KDIM + S_ + k] = f2bf(__builtin_nontemporal_load(
          inp + ((size_t)(b0 + m) * T_ + step) * IN_ + k));
      if (tid < MT * S_) {
        int mx = tid >> 1, s = tid & 1;
        u_buf[mx * KDIM + s] = f2bf(hbuf[mx][s]); // x_prev
      }
    }
    __syncthreads();

    // (b) gates = u @ Wc^T : 4 gate tiles x 18 K-steps of bf16 WMMA.
    // The weights are T-loop invariant; an opaque per-iteration offset stops
    // the compiler from hoisting all 72 fragments and spilling to scratch,
    // while keeping the pointer rooted at the kernel arg so the loads lower
    // to global_load_b128 served from the 192MB L2 (Wc is 2.25MB).
    unsigned int zofs = 0;
    asm volatile("" : "+v"(zofs));   // opaque zero: defeat LICM, keep provenance
    const unsigned short* bp0 =
        Wc + (size_t)(wave * KT) * (NTILE * 32 * 16) + lane * 16 + zofs;
    v8f acc[4];
    #pragma unroll
    for (int t = 0; t < 4; ++t) acc[t] = (v8f){0.f,0.f,0.f,0.f,0.f,0.f,0.f,0.f};
    #pragma unroll
    for (int kb = 0; kb < KT; ++kb) {
      Frag a;
      const unsigned short* ur = u_buf + q * KDIM + kb * 32;
      a.q[0] = *(const u32x4*)(const void*)(ur + hi * 16);       // K 0..7 / 16..23
      a.q[1] = *(const u32x4*)(const void*)(ur + hi * 16 + 8);   // K 8..15 / 24..31
      const unsigned short* bp = bp0 + (size_t)kb * (NTILE * 32 * 16);
      if (kb + 1 < KT)
        // WGP-scope prefetch of next K-tile; distance folded into the
        // instruction immediate so no VALU lands in the WMMA hazard shadow.
        asm volatile("global_prefetch_b8 %0, off offset:%1"
                     :: "v"(bp0), "i"((kb + 1) * WTILE_BYTES));
      #pragma unroll
      for (int t = 0; t < 4; ++t) {
        Frag b;
        b.q[0] = *(const u32x4*)(const void*)(bp + t * (32 * 16));
        b.q[1] = *(const u32x4*)(const void*)(bp + t * (32 * 16) + 8);
        acc[t] = __builtin_amdgcn_wmma_f32_16x16x32_bf16(
            false, a.v, false, b.v, (short)0, acc[t], false, false);
      }
    }

    // gate nonlinearities + cell update, fully in registers
    float hv[8];
    #pragma unroll
    for (int v = 0; v < 8; ++v) {
      float ig = sigmoidf(acc[0][v] + bias[0]);
      float fg = sigmoidf(acc[1][v] + bias[1]);
      float gg = tanhf  (acc[2][v] + bias[2]);
      float og = sigmoidf(acc[3][v] + bias[3]);
      float cn = fg * c[v] + ig * gg;
      c[v] = cn;
      hv[v] = og * tanhf(cn);
    }
    if (jv) {
      #pragma unroll
      for (int v = 0; v < 8; ++v) hbuf[hi * 8 + v][S_ + j] = hv[v];  // f32 for ODE/out
    }
    __syncthreads();

    // (c1) ODE path on waves 0..15; ALL waves refresh the bf16 h-region of
    // u_buf from their register h (safe: GEMM reads of u_buf ended at B2).
    if (jv) {
      #pragma unroll
      for (int v = 0; v < 8; ++v)
        u_buf[(hi * 8 + v) * KDIM + S_ + IN_ + j] = f2bf(hv[v]);
    }
    if (wave < MT) {
      const int m = wave;
      float a0 = 0.f, a1 = 0.f, h0 = 0.f, h1 = 0.f;
      for (int k = lane; k < H_; k += 32) {
        float h = hbuf[m][S_ + k];
        a0 += h * alpha_w[k];
        a1 += h * alpha_w[H_ + k];
        h0 += h * hx_w[k];
        h1 += h * hx_w[H_ + k];
      }
      #pragma unroll
      for (int off = 16; off > 0; off >>= 1) {
        a0 += __shfl_down(a0, off);
        a1 += __shfl_down(a1, off);
        h0 += __shfl_down(h0, off);
        h1 += __shfl_down(h1, off);
      }
      if (lane == 0) {
        float x0 = hbuf[m][0], x1 = hbuf[m][1];
        float tt = __builtin_nontemporal_load(tau + (size_t)(b0 + m) * T_ + step);
        float xm0 = x0 + tt * (MHU_ * x0 - MHU_ * x1);   // x @ A^T row 0
        float xm1 = x1 + tt * (x0 / MHU_);               // x @ A^T row 1
        float al0 = sigmoidf(a0 + alpha_b[0]);
        float al1 = sigmoidf(a1 + alpha_b[1]);
        hbuf[m][0] = (1.0f - al0) * xm0 + al0 * (h0 + hx_b[0]);
        hbuf[m][1] = (1.0f - al1) * xm1 + al1 * (h1 + hx_b[1]);
      }
    }
    __syncthreads();

    // (c2) stream full rows [x|h] to out[b, step, :] -- 16B-coalesced NT b128
    // stores: 512MB written once and never re-read, keep it out of the
    // weight-resident caches.
    for (int idx = tid; idx < MT * (Z_ / 4); idx += 1024) {
      int m  = idx >> 7;                 // Z_/4 = 128 vectors per row
      int k4 = (idx & 127) << 2;
      __builtin_nontemporal_store(
          *(const f32x4*)(const void*)&hbuf[m][k4],
          (f32x4*)(void*)(out + ((size_t)(b0 + m) * T_ + step) * Z_ + k4));
    }
  }

  __syncthreads();
  // ---- final states z_fin = [xT, hT] (block copy), cz_fin = [cx, cT] ----
  for (int idx = tid; idx < MT * (Z_ / 4); idx += 1024) {
    int m  = idx >> 7;
    int k4 = (idx & 127) << 2;
    *(f32x4*)(void*)(zfin + (size_t)(b0 + m) * Z_ + k4) =
        *(const f32x4*)(const void*)&hbuf[m][k4];
  }
  if (jv) {
    #pragma unroll
    for (int v = 0; v < 8; ++v) {
      int m = hi * 8 + v;
      czfin[(size_t)(b0 + m) * Z_ + S_ + j] = c[v];
    }
  }
}

// ---------------------------------------------------------------------------
extern "C" void kernel_launch(void* const* d_in, const int* in_sizes, int n_in,
                              void* d_out, int out_size, void* d_ws, size_t ws_size,
                              hipStream_t stream) {
  const float* rnn_input = (const float*)d_in[0];
  const float* tau       = (const float*)d_in[1];
  const float* z0        = (const float*)d_in[2];
  const float* cz0       = (const float*)d_in[3];
  const float* W_w       = (const float*)d_in[4];
  const float* W_b       = (const float*)d_in[5];
  const float* U_w       = (const float*)d_in[6];
  const float* alpha_w   = (const float*)d_in[7];
  const float* alpha_b   = (const float*)d_in[8];
  const float* hx_w      = (const float*)d_in[9];
  const float* hx_b      = (const float*)d_in[10];
  unsigned short* Wc = (unsigned short*)d_ws;    // needs 2.25 MB of workspace
  float* out = (float*)d_out;

  const int NTOT = WAVES * KT * NTILE * 32 * 16; // 1,179,648 bf16 elements
  ctrnn_pack_weights<<<dim3((NTOT + 255) / 256), dim3(256), 0, stream>>>(W_w, U_w, Wc);
  ctrnn_persistent<<<dim3(NBLK), dim3(1024), 0, stream>>>(
      rnn_input, tau, z0, cz0, W_b, alpha_w, alpha_b, hx_w, hx_b, Wc, out);
}
